// DeBERTaHead_61881888800789
// MI455X (gfx1250) — matrix operand order repaired and checked by reading, not compiled, not run on hardware
//
#include <hip/hip_runtime.h>

// DeBERTa disentangled attention head, MI455X (gfx1250, wave32, WMMA).
// B=4, T=4096, E=1024, H=64. attn output (268MB) dominates HBM traffic
// (~14us floor @ 23.3TB/s); all intermediates are L2-resident, so every
// GEMM runs as v_wmma_f32_16x16x32_bf16 (f32 accumulate), softmax in f32.
// Operands are laid out so every WMMA A/B fetch is a contiguous 16/32B
// vector load (weights and v pre-transposed).

#define BB 4
#define TT 4096
#define EE 1024
#define HH 64

typedef __bf16 v16bf __attribute__((ext_vector_type(16)));
typedef __bf16 v8bf  __attribute__((ext_vector_type(8)));
typedef __bf16 v2bf  __attribute__((ext_vector_type(2)));
typedef float  v8f   __attribute__((ext_vector_type(8)));
typedef float  v2f   __attribute__((ext_vector_type(2)));

static __device__ __forceinline__ v8f wmma_bf16(v16bf a, v16bf b, v8f c) {
  // D = A(16x32) * B(32x16) + C(16x16 f32)
  return __builtin_amdgcn_wmma_f32_16x16x32_bf16(
      false, a, false, b, (short)0, c, false, false);
}

static __device__ __forceinline__ v16bf cat8(v8bf lo, v8bf hi) {
  return __builtin_shufflevector(lo, hi, 0, 1, 2, 3, 4, 5, 6, 7,
                                 8, 9, 10, 11, 12, 13, 14, 15);
}

// A operand (16x32 bf16): lane row = M, elements [kk+half*16, +16) —
// two contiguous 16B chunks.
static __device__ __forceinline__ v16bf loadA(const __bf16* row, int kk, int half) {
  const v8bf* p = (const v8bf*)(row + kk + half * 16);
  return cat8(p[0], p[1]);
}

// B operand (32x16 bf16): lane col = N, elements K = half*16 + 0..15 —
// one contiguous 32B chunk when the source is K-contiguous per column.
static __device__ __forceinline__ v16bf loadB(const __bf16* p) {
  return *(const v16bf*)p;
}

// A operand built from f32 data (probabilities in LDS), converted to bf16.
static __device__ __forceinline__ v16bf loadA_f32(const float* row, int kk, int half) {
  v8f lo = *(const v8f*)(row + kk + half * 16);
  v8f hi = *(const v8f*)(row + kk + half * 16 + 8);
  v16bf r;
  #pragma unroll
  for (int j = 0; j < 8; ++j) { r[j] = (__bf16)lo[j]; r[8 + j] = (__bf16)hi[j]; }
  return r;
}

// ---- weight conversion: 5x (1024x64) f32 -> bf16, TRANSPOSED to [n][k] ----
__global__ __launch_bounds__(256) void kconv_w(const float* w0, const float* w1,
                                               const float* w2, const float* w3,
                                               const float* w4, __bf16* wbfT) {
  int i = blockIdx.x * 256 + threadIdx.x;
  if (i >= 5 * EE * HH) return;
  int slot = i >> 16;               // EE*HH == 65536
  int rem = i & 65535;
  int n = rem >> 10, k = rem & (EE - 1);
  const float* src = (slot == 0) ? w0 : (slot == 1) ? w1 : (slot == 2) ? w2
                     : (slot == 3) ? w3 : w4;
  wbfT[i] = (__bf16)src[k * HH + n];   // wbfT[slot][n][k] = W[k][n]
}

// ---------------- pos projections: pk (bf16), pq (f32) ----------------
__global__ __launch_bounds__(128) void kproj_pos(const float* pos, const __bf16* wbfT,
                                                 __bf16* pkbf, float* pqf) {
  const int tb  = blockIdx.x * 16;
  const int tid = threadIdx.x;
  extern __shared__ __bf16 lp[];                 // 16 x 1024 bf16 tile
  for (int i = tid; i < 16 * EE / 2; i += 128) { // packed pair stores
    int r = i >> 9, c = (i & 511) << 1;
    v2f xa = *(const v2f*)(pos + (tb + r) * EE + c);
    v2bf t; t[0] = (__bf16)xa[0]; t[1] = (__bf16)xa[1];
    *(v2bf*)(lp + r * EE + c) = t;
  }
  __syncthreads();
  const int lane = tid & 31, wave = tid >> 5;
  const int l16 = lane & 15, half = lane >> 4;
  const int n0 = wave * 16;                      // wave <-> N-tile (H/16 == 4)
  const __bf16* lrow = lp + l16 * EE;
  for (int which = 0; which < 2; ++which) {      // 0: W_pk (slot2), 1: W_pq (slot3)
    const __bf16* wrow = wbfT + (which ? 3 : 2) * (EE * HH) + (n0 + l16) * EE;
    v8f acc0 = {}, acc1 = {};
    for (int kk = 0; kk < EE; kk += 64) {
      acc0 = wmma_bf16(loadA(lrow, kk, half),
                       loadB(wrow + kk + half * 16), acc0);
      acc1 = wmma_bf16(loadA(lrow, kk + 32, half),
                       loadB(wrow + kk + 32 + half * 16), acc1);
    }
    v8f acc = acc0 + acc1;
    #pragma unroll
    for (int i = 0; i < 8; ++i) {
      int m = i + half * 8;
      int o = (tb + m) * HH + n0 + l16;
      if (which == 0) pkbf[o] = (__bf16)acc[i];
      else            pqf[o]  = acc[i];
    }
  }
}

// ------- x projections: ck, cq, q'=cq+pq (bf16); v*scale TRANSPOSED -------
__global__ __launch_bounds__(128) void kproj_x(const float* x, const __bf16* wbfT,
                                               const float* pqf, __bf16* ckbf,
                                               __bf16* cqbf, __bf16* qpbf,
                                               __bf16* vT) {
  const int tb  = blockIdx.x * 16;
  const int bb  = blockIdx.y;
  const int tid = threadIdx.x;
  const long row0 = (long)bb * TT + tb;
  extern __shared__ __bf16 lx[];                 // 16 x 1024 bf16 tile
  for (int i = tid; i < 16 * EE / 2; i += 128) {
    int r = i >> 9, c = (i & 511) << 1;
    v2f xa = *(const v2f*)(x + (row0 + r) * EE + c);
    v2bf t; t[0] = (__bf16)xa[0]; t[1] = (__bf16)xa[1];
    *(v2bf*)(lx + r * EE + c) = t;
  }
  __syncthreads();
  const int lane = tid & 31, wave = tid >> 5;
  const int l16 = lane & 15, half = lane >> 4;
  const int n0 = wave * 16;
  const int slot[3] = {0, 1, 4};                 // W_ck, W_cq, W_v
  const __bf16* lrow = lx + l16 * EE;
  for (int which = 0; which < 3; ++which) {
    const __bf16* wrow = wbfT + slot[which] * (EE * HH) + (n0 + l16) * EE;
    v8f acc0 = {}, acc1 = {};
    for (int kk = 0; kk < EE; kk += 64) {
      acc0 = wmma_bf16(loadA(lrow, kk, half),
                       loadB(wrow + kk + half * 16), acc0);
      acc1 = wmma_bf16(loadA(lrow, kk + 32, half),
                       loadB(wrow + kk + 32 + half * 16), acc1);
    }
    v8f acc = acc0 + acc1;
    #pragma unroll
    for (int i = 0; i < 8; ++i) {
      int m = i + half * 8;
      long t = row0 + m;
      int col = n0 + l16;
      float val = acc[i];
      if (which == 0) {
        ckbf[t * HH + col] = (__bf16)val;
      } else if (which == 1) {
        cqbf[t * HH + col] = (__bf16)val;
        qpbf[t * HH + col] = (__bf16)(val + pqf[(tb + m) * HH + col]);
      } else {
        // vT[b][h][t]: K-contiguous for the p@v B operand
        vT[((long)bb * HH + col) * TT + tb + m] =
            (__bf16)(val * 0.07216878364870323f);   // (3*64)^-0.5
      }
    }
  }
}

// ---------------- attention: scores -> softmax -> attn + out ----------------
__global__ __launch_bounds__(256) void kattn(const __bf16* qpbf, const __bf16* cqbf,
                                             const __bf16* ckbf, const __bf16* pkbf,
                                             const __bf16* vT, float* attn,
                                             float* outp) {
  const int qt = blockIdx.x, bb = blockIdx.y;
  const int qb = qt * 16;
  const int tid = threadIdx.x;
  const int lane = tid & 31, wave = tid >> 5;
  const int l16 = lane & 15, half = lane >> 4;

  extern __shared__ float sm[];
  float* S      = sm;                 // 65536 floats (16 x 4096 scores)
  float* red    = sm + 65536;         // 256
  float* rowmax = red + 256;          // 16
  float* rowinv = rowmax + 16;        // 16
  float* opart  = rowinv + 16;        // 8*256

  const float NEGINF = -__builtin_huge_valf();
  for (int i = tid; i < 16 * TT; i += 256) S[i] = NEGINF;
  __syncthreads();

  const long rq = (long)bb * TT + qb;

  // Hoist A operands (q' and cq rows) — identical across all k-tiles.
  v16bf aqp[2], acq[2];
  {
    const __bf16* q1 = qpbf + (rq + l16) * HH;
    const __bf16* q2 = cqbf + (rq + l16) * HH;
    #pragma unroll
    for (int s = 0; s < 2; ++s) {
      aqp[s] = loadA(q1, s * 32, half);
      acq[s] = loadA(q2, s * 32, half);
    }
  }

  // --- score tiles (causal: only kt <= qt), wave-strided, 2-way ILP ---
  #pragma unroll 2
  for (int kt = wave; kt <= qt; kt += 8) {
    const int kb = kt * 16;
    const __bf16* kr = ckbf + ((long)bb * TT + kb + l16) * HH;  // B[k][n]=ck[n][k]
    const __bf16* pr = pkbf + (long)(kb + l16) * HH;
    v8f acc = {};
    #pragma unroll
    for (int s = 0; s < 2; ++s) {
      const int kbk = s * 32 + half * 16;
      acc = wmma_bf16(aqp[s], loadB(kr + kbk), acc);
      acc = wmma_bf16(acq[s], loadB(pr + kbk), acc);
    }
    #pragma unroll
    for (int i = 0; i < 8; ++i) {
      int m = i + half * 8;
      int kidx = kb + l16;
      S[m * TT + kidx] = (kidx <= qb + m) ? acc[i] : NEGINF;
    }
  }
  __syncthreads();

  // --- row max / sum (16 threads per row) ---
  const int row = tid >> 4, sub = tid & 15;
  const int limit = qb + row;
  float lmax = NEGINF;
  for (int k = sub; k <= limit; k += 16) lmax = fmaxf(lmax, S[row * TT + k]);
  red[row * 16 + sub] = lmax;
  __syncthreads();
  if (sub == 0) {
    float m = red[row * 16];
    #pragma unroll
    for (int j = 1; j < 16; ++j) m = fmaxf(m, red[row * 16 + j]);
    rowmax[row] = m;
  }
  __syncthreads();
  const float rm = rowmax[row];
  float lsum = 0.f;
  for (int k = sub; k <= limit; k += 16) lsum += __expf(S[row * TT + k] - rm);
  red[row * 16 + sub] = lsum;
  __syncthreads();
  if (sub == 0) {
    float s = 0.f;
    #pragma unroll
    for (int j = 0; j < 16; ++j) s += red[row * 16 + j];
    rowinv[row] = 1.0f / s;
  }
  __syncthreads();

  // --- normalize: write attn to HBM (the only big store), keep p in LDS ---
  for (int idx = tid; idx < 16 * TT; idx += 256) {
    int m = idx >> 12, k = idx & (TT - 1);
    float p = __expf(S[idx] - rowmax[m]) * rowinv[m];   // exp(-inf) == 0
    S[idx] = p;
    attn[(size_t)bb * TT * TT + (size_t)(qb + m) * TT + k] = p;
  }
  __syncthreads();

  // --- out = p @ v : waves split 4 N-tiles x 2 K-halves, dual accumulators ---
  {
    const int ntile = wave & 3, khalf = wave >> 2;
    const int n0 = ntile * 16;
    const float* Srow = S + l16 * TT;
    const __bf16* vrow = vT + ((long)bb * HH + n0 + l16) * TT;  // K-contiguous
    v8f acc0 = {}, acc1 = {};
    const int kbase = khalf * 2048;
    for (int kb = kbase; kb < kbase + 2048; kb += 64) {
      acc0 = wmma_bf16(loadA_f32(Srow, kb, half),
                       loadB(vrow + kb + half * 16), acc0);
      acc1 = wmma_bf16(loadA_f32(Srow, kb + 32, half),
                       loadB(vrow + kb + 32 + half * 16), acc1);
    }
    v8f acc = acc0 + acc1;
    #pragma unroll
    for (int i = 0; i < 8; ++i)
      opart[wave * 256 + (i + half * 8) * 16 + l16] = acc[i];
  }
  __syncthreads();
  for (int idx = tid; idx < 16 * HH; idx += 256) {
    int m = idx >> 6, h = idx & 63;
    int nt = h >> 4, n = h & 15;
    float val = opart[nt * 256 + m * 16 + n] + opart[(nt + 4) * 256 + m * 16 + n];
    outp[((size_t)bb * TT + qb + m) * HH + h] = val;
  }
}

// ---------------- launch ----------------
extern "C" void kernel_launch(void* const* d_in, const int* in_sizes, int n_in,
                              void* d_out, int out_size, void* d_ws, size_t ws_size,
                              hipStream_t stream) {
  const float* x   = (const float*)d_in[0];
  const float* pos = (const float*)d_in[1];
  const float* Wck = (const float*)d_in[2];
  const float* Wcq = (const float*)d_in[3];
  const float* Wpk = (const float*)d_in[4];
  const float* Wpq = (const float*)d_in[5];
  const float* Wv  = (const float*)d_in[6];

  char* ws = (char*)d_ws;                       // ~10.6 MB used
  __bf16* wbfT = (__bf16*)(ws + 0);             // 5 * 64 * 1024 bf16 (transposed)
  __bf16* pkbf = (__bf16*)(ws + 655360);        // T*H bf16
  float*  pqf  = (float*) (ws + 1179648);       // T*H f32
  __bf16* ckbf = (__bf16*)(ws + 2228224);       // B*T*H bf16
  __bf16* cqbf = (__bf16*)(ws + 4325376);
  __bf16* qpbf = (__bf16*)(ws + 6422528);
  __bf16* vT   = (__bf16*)(ws + 8519680);       // B*H*T bf16 (transposed)

  float* attn = (float*)d_out;
  float* outp = attn + (size_t)BB * TT * TT;

  kconv_w<<<dim3((5 * EE * HH + 255) / 256), dim3(256), 0, stream>>>(
      Wck, Wcq, Wpk, Wpq, Wv, wbfT);
  kproj_pos<<<dim3(TT / 16), dim3(128), 16 * EE * sizeof(__bf16), stream>>>(
      pos, wbfT, pkbf, pqf);
  kproj_x<<<dim3(TT / 16, BB), dim3(128), 16 * EE * sizeof(__bf16), stream>>>(
      x, wbfT, pqf, ckbf, cqbf, qpbf, vT);
  kattn<<<dim3(TT / 16, BB), dim3(256), 67872 * sizeof(float), stream>>>(
      qpbf, cqbf, ckbf, pkbf, vT, attn, outp);
}
